// ScaledDotPorductAttention_11467562680632
// MI455X (gfx1250) — matrix operand order repaired and checked by reading, not compiled, not run on hardware
//
#include <hip/hip_runtime.h>

typedef __attribute__((ext_vector_type(16))) __bf16 v16bf;
typedef __attribute__((ext_vector_type(8)))  __bf16 v8bf;
typedef __attribute__((ext_vector_type(8)))  float  v8f;

static constexpr int Bx = 2;     // batch
static constexpr int Hh = 16;    // heads
static constexpr int Ss = 2048;  // sequence
static constexpr int Dd = 128;   // head dim
static constexpr int BM = 128;   // query rows per workgroup (16 per wave x 8 waves)
static constexpr int BN = 64;    // kv rows per tile
static constexpr int NT = Ss / BN;  // 32 kv tiles
static constexpr int KSTR = 136; // lds K row stride (bf16): 272B, 16B-aligned, bank-skewed
static constexpr int VSTR = 72;  // lds V^T row stride (bf16): 144B, 16B-aligned
static constexpr int PSTR = 72;  // lds P row stride

__device__ __forceinline__ v16bf cat8(v8bf a, v8bf b) {
  return __builtin_shufflevector(a, b, 0,1,2,3,4,5,6,7,8,9,10,11,12,13,14,15);
}

__device__ __forceinline__ v8f wmma_bf16(v16bf a, v16bf b, v8f c) {
  // D = A(16x32 bf16) x B(32x16 bf16) + C(f32)
  return __builtin_amdgcn_wmma_f32_16x16x32_bf16(false, a, false, b, (short)0, c,
                                                 false, false);
}

__global__ __launch_bounds__(256, 1)
void fa_fwd_bf16wmma(const float* __restrict__ Q, const float* __restrict__ K,
                     const float* __restrict__ V, float* __restrict__ O)
{
  // double-buffered K / V^T tiles + per-wave P staging
  __shared__ __bf16 lds_k[2 * BN * KSTR];      // [buf][key][dim]
  __shared__ __bf16 lds_vt[2 * Dd * VSTR];     // [buf][dim][key]
  __shared__ __bf16 lds_p[8 * 16 * PSTR];      // [wave][m][kv]

  const int qb   = blockIdx.x;         // query block 0..15
  const int h    = blockIdx.y;
  const int b    = blockIdx.z;
  const int tid  = threadIdx.x;
  const int wave = tid >> 5;
  const int lane = tid & 31;
  const int nl   = lane & 15;          // column-within-tile lane id
  const int hi   = lane >> 4;          // half-wave select

  const size_t head_off = ((size_t)(b * Hh + h)) * Ss * Dd;
  const float* q = Q + head_off;
  const float* k = K + head_off;
  const float* v = V + head_off;
  float*       o = O + head_off;

  // softmax runs in exp2 domain: fold 1/sqrt(128) * log2(e) into the score scale
  const float scale2 = 0.08838834764831845f * 1.44269504088896341f;

  // per-thread staging coordinates (same for every tile)
  int srow[8], scol[8];
#pragma unroll
  for (int i = 0; i < 8; ++i) {
    const int idx = tid + i * 256;
    srow[i] = idx >> 5;            // 0..63
    scol[i] = (idx & 31) << 2;     // 0..124 step 4
  }

  // ---- Load this wave's Q slab (16 x 128) straight into A-layout bf16 frags.
  // A(16x32,16-bit): lane m=nl; elems 0..7 -> K = 8*hi + j ; elems 8..15 -> K = 16+8*hi+j
  const int qbase = qb * BM + wave * 16;
  v16bf aq[4];
  {
    const float* qrow = q + (size_t)(qbase + nl) * Dd;
#pragma unroll
    for (int c = 0; c < 4; ++c) {
      const float* p0 = qrow + 32 * c + 8 * hi;
      const float* p1 = qrow + 32 * c + 16 + 8 * hi;
#pragma unroll
      for (int j = 0; j < 8; ++j) {
        aq[c][j]     = (__bf16)p0[j];
        aq[c][8 + j] = (__bf16)p1[j];
      }
    }
  }

  // ---- Online-softmax state. C/D layout: vgpr r <-> row m = r + 8*hi.
  v8f acc[8];                       // O accumulator: 8 dim-tiles of 16x16 f32
#pragma unroll
  for (int t = 0; t < 8; ++t) acc[t] = v8f{};
  float mrow[8], lrow[8];
#pragma unroll
  for (int r = 0; r < 8; ++r) { mrow[r] = -1e30f; lrow[r] = 0.0f; }

  // ---- Prologue: pull tile 0 into registers (32-bit offsets -> saddr+voffset).
  float4 kreg[8], vreg[8];
#pragma unroll
  for (int i = 0; i < 8; ++i) {
    const int off = srow[i] * Dd + scol[i];
    kreg[i] = *reinterpret_cast<const float4*>(k + off);
    vreg[i] = *reinterpret_cast<const float4*>(v + off);
  }

  for (int it = 0; it < NT; ++it) {
    const int buf = it & 1;
    __bf16* lk = lds_k  + buf * (BN * KSTR);
    __bf16* lv = lds_vt + buf * (Dd * VSTR);

    // ---- Drain staged registers into this iteration's LDS buffer (cvt to bf16).
#pragma unroll
    for (int i = 0; i < 8; ++i) {
      __bf16* kd = &lk[srow[i] * KSTR + scol[i]];
      kd[0] = (__bf16)kreg[i].x; kd[1] = (__bf16)kreg[i].y;
      kd[2] = (__bf16)kreg[i].z; kd[3] = (__bf16)kreg[i].w;
      lv[(scol[i] + 0) * VSTR + srow[i]] = (__bf16)vreg[i].x;
      lv[(scol[i] + 1) * VSTR + srow[i]] = (__bf16)vreg[i].y;
      lv[(scol[i] + 2) * VSTR + srow[i]] = (__bf16)vreg[i].z;
      lv[(scol[i] + 3) * VSTR + srow[i]] = (__bf16)vreg[i].w;
    }
    __syncthreads();

    // ---- Issue next tile's global loads now; they drain under this tile's WMMAs.
    if (it + 1 < NT) {
      const int tbase = (it + 1) * BN * Dd;
#pragma unroll
      for (int i = 0; i < 8; ++i) {
        const int off = tbase + srow[i] * Dd + scol[i];
        kreg[i] = *reinterpret_cast<const float4*>(k + off);
        vreg[i] = *reinterpret_cast<const float4*>(v + off);
      }
      if (it + 2 < NT) {  // global_prefetch_b8 hint two tiles ahead
        const int pbase = (it + 2) * BN * Dd + tid * 32;
        __builtin_prefetch(k + pbase, 0, 1);
        __builtin_prefetch(v + pbase, 0, 1);
      }
    }

    // ---- S = Q K^T (4 key tiles x 4 k-chunks of 32).
    // B(32x16,16-bit): lane n=nl; elems 0..7 -> K(dim) = 8*hi + j ; elems 8..15 -> K = 16+8*hi+j
    v8f s[4];
#pragma unroll
    for (int t = 0; t < 4; ++t) {
      v8f c = v8f{};
#pragma unroll
      for (int cc = 0; cc < 4; ++cc) {
        const __bf16* kr = &lk[(16 * t + nl) * KSTR + 32 * cc + 8 * hi];
        const v16bf bk = cat8(*(const v8bf*)kr, *(const v8bf*)(kr + 16));
        c = wmma_bf16(aq[cc], bk, c);
      }
      s[t] = c;
    }
#pragma unroll
    for (int t = 0; t < 4; ++t)
#pragma unroll
      for (int r = 0; r < 8; ++r) s[t][r] *= scale2;   // now in log2 units

    // ---- Online softmax (exp2 domain): row max / rescale / exp2 / row sum.
    float alpha[8];
#pragma unroll
    for (int r = 0; r < 8; ++r) {
      float mx = fmaxf(fmaxf(s[0][r], s[1][r]), fmaxf(s[2][r], s[3][r]));
#pragma unroll
      for (int off = 1; off < 16; off <<= 1)
        mx = fmaxf(mx, __shfl_xor(mx, off, 32));   // reduce within 16-lane half
      const float mn = fmaxf(mrow[r], mx);
      alpha[r] = exp2f(mrow[r] - mn);
      mrow[r] = mn;
    }

    float psum[8];
#pragma unroll
    for (int r = 0; r < 8; ++r) psum[r] = 0.0f;
    __bf16* pbuf = &lds_p[wave * 16 * PSTR];
#pragma unroll
    for (int t = 0; t < 4; ++t) {
#pragma unroll
      for (int r = 0; r < 8; ++r) {
        const float p = exp2f(s[t][r] - mrow[r]);
        psum[r] += p;
        pbuf[(r + 8 * hi) * PSTR + 16 * t + nl] = (__bf16)p;  // C-layout -> LDS
      }
    }
#pragma unroll
    for (int r = 0; r < 8; ++r) {
      float ps = psum[r];
#pragma unroll
      for (int off = 1; off < 16; off <<= 1) ps += __shfl_xor(ps, off, 32);
      lrow[r] = lrow[r] * alpha[r] + ps;
    }
#pragma unroll
    for (int t = 0; t < 8; ++t)
#pragma unroll
      for (int r = 0; r < 8; ++r) acc[t][r] *= alpha[r];

    // wave-private P buffer: only need our own LDS stores visible
    asm volatile("s_wait_dscnt 0" ::: "memory");

    // ---- Reload P in A-layout (two contiguous 8-elem runs per chunk).
    v16bf ap[2];
#pragma unroll
    for (int c2 = 0; c2 < 2; ++c2) {
      const __bf16* pr = &pbuf[nl * PSTR + 32 * c2 + 8 * hi];
      ap[c2] = cat8(*(const v8bf*)pr, *(const v8bf*)(pr + 16));
    }

    // ---- O += P V  (8 dim tiles x 2 kv-chunks of 32).
#pragma unroll
    for (int t = 0; t < 8; ++t) {
      v8f c = acc[t];
#pragma unroll
      for (int c2 = 0; c2 < 2; ++c2) {
        const __bf16* vr = &lv[(16 * t + nl) * VSTR + 32 * c2 + 8 * hi];
        const v16bf bv = cat8(*(const v8bf*)vr, *(const v8bf*)(vr + 16));
        c = wmma_bf16(ap[c2], bv, c);
      }
      acc[t] = c;
    }
  }

  // ---- Normalize and write O (f32).
  float invl[8];
#pragma unroll
  for (int r = 0; r < 8; ++r) invl[r] = 1.0f / lrow[r];
#pragma unroll
  for (int t = 0; t < 8; ++t)
#pragma unroll
    for (int r = 0; r < 8; ++r)
      o[(size_t)(qbase + r + 8 * hi) * Dd + 16 * t + nl] = acc[t][r] * invl[r];
}

extern "C" void kernel_launch(void* const* d_in, const int* in_sizes, int n_in,
                              void* d_out, int out_size, void* d_ws, size_t ws_size,
                              hipStream_t stream) {
  const float* q = (const float*)d_in[0];
  const float* k = (const float*)d_in[1];
  const float* v = (const float*)d_in[2];
  float* out = (float*)d_out;
  dim3 grid(Ss / BM, Hh, Bx);  // (16, 16, 2)
  dim3 block(256);
  fa_fwd_bf16wmma<<<grid, block, 0, stream>>>(q, k, v, out);
}
